// FwFM_83897891160139
// MI455X (gfx1250) — matrix operand order repaired and checked by reading, not compiled, not run on hardware
//
#include <hip/hip_runtime.h>
#include <hip/hip_bf16.h>
#include <stdint.h>

// FwFM: F=26 fields, B=16384 batch, L=8 bag, V=100000 vocab, D=64 dim.
#define NF 26
#define NB 16384
#define NL 8
#define NV 100000
#define ND 64
#define FP 32 // fields padded to 32 for 16x16 WMMA tiling

typedef __attribute__((ext_vector_type(16))) __bf16 v16bf;
typedef __attribute__((ext_vector_type(8)))  float  v8f;

// round-to-nearest-even f32 -> bf16, pack two into a dword
__device__ __forceinline__ uint32_t pack_bf16(float a, float b) {
    uint32_t ua = __float_as_uint(a);
    uint32_t ub = __float_as_uint(b);
    ua += 0x7FFFu + ((ua >> 16) & 1u);
    ub += 0x7FFFu + ((ub >> 16) & 1u);
    return (ua >> 16) | (ub & 0xFFFF0000u);
}

// ---------------------------------------------------------------------------
// Kernel A: weighted embedding bag. One wave per (field,b). Lane i owns
// d = {2i, 2i+1}. Indices/weights are wave-uniform -> forced into SGPRs so
// the gather is scalar-addressed global_load_b64 per embedding row.
// Writes vecs as packed bf16 [B][32][64] (rows 26..31 zeroed) and the linear
// partial dot into lin_part[B][32] (padded slots zeroed).
// ---------------------------------------------------------------------------
__global__ __launch_bounds__(256) void fwfm_bag_kernel(
    const int* __restrict__ idxs, const float* __restrict__ weights,
    const float* __restrict__ emb, const float* __restrict__ lin_w,
    uint32_t* __restrict__ vecs, float* __restrict__ lin_part)
{
    const int wave = blockIdx.x * 8 + (threadIdx.x >> 5);
    const int lane = threadIdx.x & 31;
    const int f = wave & (FP - 1);
    const int b = wave >> 5;

    float2 acc = make_float2(0.f, 0.f);
    float  lp  = 0.f;
    if (f < NF) {
        const long base = ((long)f * NB + b) * NL;
        #pragma unroll
        for (int l = 0; l < NL; ++l) {
            int   idx = __builtin_amdgcn_readfirstlane(idxs[base + l]);
            float w   = __uint_as_float(
                __builtin_amdgcn_readfirstlane(__float_as_uint(weights[base + l])));
            const float2* row = (const float2*)(emb + ((long)f * NV + idx) * ND);
            float2 e = row[lane];
            acc.x = fmaf(w, e.x, acc.x);
            acc.y = fmaf(w, e.y, acc.y);
        }
        // linear partial: vecs[b,f]·lin_w[f]
        const float2* lw = (const float2*)(lin_w + f * ND);
        float2 l2 = lw[lane];
        lp = acc.x * l2.x + acc.y * l2.y;
        #pragma unroll
        for (int off = 16; off >= 1; off >>= 1) lp += __shfl_xor(lp, off, 32);
    }
    if (lane == 0) lin_part[(long)b * FP + f] = lp; // 0 for padded fields
    // store packed bf16 row (f >= NF stores zeros: workspace is poisoned)
    vecs[((long)b * FP + f) * (ND / 2) + lane] = pack_bf16(acc.x, acc.y);
}

// ---------------------------------------------------------------------------
// Kernel B: one wave per batch element. Gram G = V·V^T (32x32, K=64) with
// v_wmma_f32_16x16x32_bf16. Tile (m=1,n=0) is entirely below the strict
// upper triangle -> skipped: 3 C-tiles x 2 K-steps = 6 WMMAs per batch.
// out[b] = bias + sum_f lin_part + sum_{f<g} r[f,g] * G[f,g].
//
// rt' is materialized in LDS once per block, padded 32x32, masked (zeros off
// the strict upper triangle / padding), and TRANSPOSED (s_rt[fN][fM]) so each
// lane's 8 tile weights are contiguous -> unconditional ds_load_b128 pairs.
//
// Fragment assembly (contiguous K order in A and B, wave32):
//   A 16x32 bf16: lane r (r=lane&15) holds row M=r; lane half selects
//     K = 0..15 (half 0) or 16..31 (half 1), dword-packed.
//   B 32x16 bf16: lane r holds col N=r; VGPR j holds K-pair {2j,2j+1}
//     (half 0: K 0..15, half 1: K 16..31).  B[k][n] = V[n][k].
//   C 16x16 f32: VGPR i -> M = i + 8*half, N = lane&15.
// ---------------------------------------------------------------------------
__global__ __launch_bounds__(256) void fwfm_wmma_kernel(
    const uint32_t* __restrict__ vecs, const float* __restrict__ lin_part,
    const float* __restrict__ r, const float* __restrict__ bias,
    float* __restrict__ out)
{
    __shared__ float s_rt[FP * FP]; // transposed + masked: s_rt[fN*32 + fM]
    for (int i = threadIdx.x; i < FP * FP; i += 256) {
        const int fN = i >> 5;
        const int fM = i & 31;
        s_rt[i] = (fM < NF && fN < NF && fN > fM) ? r[fM * NF + fN] : 0.f;
    }
    __syncthreads();

    const int b    = blockIdx.x * 8 + (threadIdx.x >> 5);
    const int lane = threadIdx.x & 31;
    const int half = lane >> 4;
    const int lrow = lane & 15;

    const uint32_t* Vb = vecs + (size_t)b * FP * (ND / 2); // 32 rows x 32 dwords

    const v8f vzero = {0.f, 0.f, 0.f, 0.f, 0.f, 0.f, 0.f, 0.f};
    v8f acc00 = vzero, acc01 = vzero, acc11 = vzero;

    #pragma unroll
    for (int ks = 0; ks < 2; ++ks) {
        const int kd = ks * 16; // dword base of this K-step (K0 = 32*ks)
        union Frag { uint32_t u[8]; v16bf v; };
        Frag Af[2], Bf[2];
        #pragma unroll
        for (int m = 0; m < 2; ++m) {
            const uint32_t* row = Vb + (size_t)(m * 16 + lrow) * (ND / 2);
            const int baseA = kd + half * 8;
            #pragma unroll
            for (int j = 0; j < 4; ++j) {
                Af[m].u[j]     = row[baseA + j];     // K = k0 + 2j, 2j+1  (+ half*16)
                Af[m].u[4 + j] = row[baseA + 4 + j]; // K = k0 + 8 + 2j  (+ half*16)
            }
        }
        #pragma unroll
        for (int n = 0; n < 2; ++n) {
            const uint32_t* row = Vb + (size_t)(n * 16 + lrow) * (ND / 2);
            const int baseB = kd + half * 8;
            #pragma unroll
            for (int j = 0; j < 8; ++j) Bf[n].u[j] = row[baseB + j];
        }
        acc00 = __builtin_amdgcn_wmma_f32_16x16x32_bf16(
                    false, Af[0].v, false, Bf[0].v, (short)0, acc00, false, false);
        acc01 = __builtin_amdgcn_wmma_f32_16x16x32_bf16(
                    false, Af[0].v, false, Bf[1].v, (short)0, acc01, false, false);
        acc11 = __builtin_amdgcn_wmma_f32_16x16x32_bf16(
                    false, Af[1].v, false, Bf[1].v, (short)0, acc11, false, false);
    }

    // weighted contraction with rt' (all-zero tile (1,0) skipped)
    float s = lin_part[(size_t)b * FP + lane]; // fold linear term (padded->0)
    auto tile = [&](const v8f& a, int m, int n) {
        const float* p = s_rt + (size_t)(n * 16 + lrow) * FP + m * 16 + half * 8;
        const float4 w0 = *(const float4*)(p);
        const float4 w1 = *(const float4*)(p + 4);
        s = fmaf(w0.x, a[0], s); s = fmaf(w0.y, a[1], s);
        s = fmaf(w0.z, a[2], s); s = fmaf(w0.w, a[3], s);
        s = fmaf(w1.x, a[4], s); s = fmaf(w1.y, a[5], s);
        s = fmaf(w1.z, a[6], s); s = fmaf(w1.w, a[7], s);
    };
    tile(acc00, 0, 0);
    tile(acc01, 0, 1);
    tile(acc11, 1, 1);

    // cross-lane reduce (wave32), single writer
    #pragma unroll
    for (int off = 16; off >= 1; off >>= 1) s += __shfl_xor(s, off, 32);
    if (lane == 0) out[b] = bias[0] + s;
}

extern "C" void kernel_launch(void* const* d_in, const int* in_sizes, int n_in,
                              void* d_out, int out_size, void* d_ws, size_t ws_size,
                              hipStream_t stream) {
    const int*   idxs    = (const int*)d_in[0];
    const float* weights = (const float*)d_in[1];
    const float* emb     = (const float*)d_in[2];
    const float* lin_w   = (const float*)d_in[3];
    const float* r       = (const float*)d_in[4];
    const float* bias    = (const float*)d_in[5];
    float* out = (float*)d_out;

    uint32_t* vecs     = (uint32_t*)d_ws;                           // B*32*32 dwords = 64MB
    float*    lin_part = (float*)((char*)d_ws +
                           (size_t)NB * FP * (ND / 2) * sizeof(uint32_t)); // B*32 f32

    // 32 field-slots * 16384 batches, 8 waves/block
    fwfm_bag_kernel<<<(FP * NB) / 8, 256, 0, stream>>>(idxs, weights, emb, lin_w,
                                                       vecs, lin_part);
    // one wave per batch element
    fwfm_wmma_kernel<<<NB / 8, 256, 0, stream>>>(vecs, lin_part, r, bias, out);
}
